// SelfAttention_69312182223600
// MI455X (gfx1250) — matrix operand order repaired and checked by reading, not compiled, not run on hardware
//
#include <hip/hip_runtime.h>

// ---------------------------------------------------------------------------
// CDNA5 (gfx1250) self-attention: prep (f16 convert + tiled weight transpose)
// -> QKV GEMM -> causal flash attention -> output GEMM.
// All matmuls via v_wmma_f32_16x16x32_f16 (wave32).  Every WMMA operand is
// stored reduction-dim-innermost so fragments load as aligned b128s.
// ---------------------------------------------------------------------------

typedef __attribute__((ext_vector_type(16))) _Float16 v16h;
typedef __attribute__((ext_vector_type(8)))  _Float16 v8h;
typedef __attribute__((ext_vector_type(4)))  _Float16 v4h;
typedef __attribute__((ext_vector_type(8)))  float    v8f;

#define B_    8
#define T_    1024
#define C_    768
#define H_    12
#define HD_   64
#define N_QKV 2304   // 3*C

static __device__ __forceinline__ v8f wmma_f16(v16h a, v16h b, v8f c) {
  // D = A(16x32 f16) * B(32x16 f16) + C(16x16 f32)
  return __builtin_amdgcn_wmma_f32_16x16x32_f16(
      false, a, false, b, (short)0, c, false, false);
}

// A-fragment: element j of lane = row[16*hi + j] (j<8 from p0, j>=8 from p1 = p0 + 8)
// -> two contiguous 8-half (16B) runs of one row.
static __device__ __forceinline__ v16h ld_afrag(const _Float16* p0,
                                                const _Float16* p1) {
  v8h x = *(const v8h*)p0;
  v8h y = *(const v8h*)p1;
  v16h r;
#pragma unroll
  for (int j = 0; j < 8; ++j) { r[j] = x[j]; r[j + 8] = y[j]; }
  return r;
}

// B-fragment: element j of lane = col[16*hi + j] -> one contiguous 32B run.
static __device__ __forceinline__ v16h ld_bfrag(const _Float16* p) {
  return *(const v16h*)p;   // 32B-aligned by construction
}

// ---------------------------------------------------------------------------
// Prep: x -> f16 (A-ready, K innermost).
// ---------------------------------------------------------------------------
__global__ __launch_bounds__(256)
void cvt_f32_f16_kernel(const float* __restrict__ in, _Float16* __restrict__ out,
                        int n4) {
  int i = blockIdx.x * 256 + threadIdx.x;
  if (i < n4) {
    float4 v = ((const float4*)in)[i];
    v4h o;
    o[0] = (_Float16)v.x; o[1] = (_Float16)v.y;
    o[2] = (_Float16)v.z; o[3] = (_Float16)v.w;
    ((v4h*)out)[i] = o;
  }
}

// ---------------------------------------------------------------------------
// Prep: W[K,N] f32 -> Wt[N,K] f16 via LDS 32x32 tiles.
// Coalesced 128B reads, coalesced 64B f16 writes, no integer division.
// grid = (N/32, K/32), block = 256.
// ---------------------------------------------------------------------------
__global__ __launch_bounds__(256)
void transpose_cvt_kernel(const float* __restrict__ W, _Float16* __restrict__ Wt,
                          int K, int N) {
  __shared__ _Float16 tile[32][33];          // pad -> conflict-free transpose
  const int n0 = blockIdx.x * 32;
  const int k0 = blockIdx.y * 32;
  const int c  = threadIdx.x & 31;
  const int r4 = threadIdx.x >> 5;           // 0..7

#pragma unroll
  for (int i = 0; i < 4; ++i) {
    int r = r4 + i * 8;                      // 0..31
    tile[r][c] = (_Float16)W[(size_t)(k0 + r) * N + (n0 + c)];
  }
  __syncthreads();
#pragma unroll
  for (int i = 0; i < 4; ++i) {
    int r = r4 + i * 8;                      // row of Wt = n index
    Wt[(size_t)(n0 + r) * K + (k0 + c)] = tile[c][r];
  }
}

// ---------------------------------------------------------------------------
// Kernel 1: QKV projection.  X16[8192x768] @ Wt[2304x768]^T + b.
// No LDS: A frags direct from X16 rows, B frags direct from transposed W.
// Scatter f16 into Q[B,H,T,HD] (x0.125), K[B,H,T,HD], V^T[B,H,HD,T].
// ---------------------------------------------------------------------------
__global__ __launch_bounds__(128)
void qkv_gemm_kernel(const _Float16* __restrict__ X16,
                     const _Float16* __restrict__ Wt,
                     const float* __restrict__ bias,
                     _Float16* __restrict__ Qo, _Float16* __restrict__ Ko,
                     _Float16* __restrict__ Vt) {
  const int lane = threadIdx.x & 31;
  const int wm   = threadIdx.x >> 5;
  const int ln16 = lane & 15;
  const int hi   = lane >> 4;
  const int n0   = blockIdx.x * 64;
  const int m0   = blockIdx.y * 64;

  const _Float16* arow = X16 + (size_t)(m0 + wm * 16 + ln16) * C_;
  v8f acc[4] = {};

  for (int k0 = 0; k0 < C_; k0 += 32) {
    v16h a = ld_afrag(arow + k0 + 16 * hi, arow + k0 + 16 * hi + 8);
#pragma unroll
    for (int nt = 0; nt < 4; ++nt) {
      v16h b = ld_bfrag(Wt + (size_t)(n0 + nt * 16 + ln16) * C_ + k0 + 16 * hi);
      acc[nt] = wmma_f16(a, b, acc[nt]);
    }
  }

#pragma unroll
  for (int nt = 0; nt < 4; ++nt) {
#pragma unroll
    for (int r = 0; r < 8; ++r) {
      int m = m0 + wm * 16 + r + hi * 8;
      int c = n0 + nt * 16 + ln16;
      float v = acc[nt][r] + bias[c];
      int b  = m >> 10;                 // / T
      int t  = m & (T_ - 1);
      int hh = c >> 6;                  // / HD : 0..35
      int which = hh / H_;              // 0=q 1=k 2=v
      int h  = hh % H_;
      int hd = c & (HD_ - 1);
      int bh = b * H_ + h;
      if (which == 0)
        Qo[((size_t)bh * T_ + t) * HD_ + hd] = (_Float16)(v * 0.125f);
      else if (which == 1)
        Ko[((size_t)bh * T_ + t) * HD_ + hd] = (_Float16)v;
      else
        Vt[((size_t)bh * HD_ + hd) * T_ + t] = (_Float16)v;   // transposed
    }
  }
}

// ---------------------------------------------------------------------------
// Kernel 2: causal flash attention.  One wave = one 16-row q tile.
// Q/K frags contiguous from [B,H,T,HD]; V frags contiguous from V^T.
// Only LDS use: P tile C-layout -> A-layout bounce (b16 stores in, b128 loads out).
// ---------------------------------------------------------------------------
__global__ __launch_bounds__(128)
void attn_kernel(const _Float16* __restrict__ Q, const _Float16* __restrict__ Kk,
                 const _Float16* __restrict__ Vt, _Float16* __restrict__ Oh) {
  __shared__ _Float16 Ps[4][16][32];        // per-wave P staging

  const int lane = threadIdx.x & 31;
  const int w    = threadIdx.x >> 5;
  const int ln16 = lane & 15;
  const int hi   = lane >> 4;

  const int tile = blockIdx.x * 4 + w;
  const int bh   = tile >> 6;               // / (T/16)
  const int q0   = (tile & 63) << 4;
  const int b    = bh / H_;
  const int h    = bh % H_;
  const size_t base = (size_t)bh * T_ * HD_;   // same extent for Q, K, V^T

  // Q fragments (two 16x32 A-frags, persist in VGPRs)
  v16h aQ[2];
  {
    const _Float16* qrow = Q + base + (size_t)(q0 + ln16) * HD_;
#pragma unroll
    for (int s = 0; s < 2; ++s)
      aQ[s] = ld_afrag(qrow + s * 32 + 16 * hi, qrow + s * 32 + 16 * hi + 8);
  }

  v8f o[4] = {};
  float mst[8], lst[8];
#pragma unroll
  for (int r = 0; r < 8; ++r) { mst[r] = -1e30f; lst[r] = 0.0f; }

  const int qend = q0 + 15;
  for (int k0 = 0; k0 <= qend; k0 += 32) {  // wave-uniform trip count
    if (k0 + 32 <= qend) {                  // prefetch next K/V chunk
      __builtin_prefetch(Kk + base + (size_t)(k0 + 32 + ln16) * HD_, 0, 0);
      __builtin_prefetch(Vt + base + (size_t)(ln16 * 4) * T_ + k0 + 32, 0, 0);
    }

    // ---- S = Q @ K^T : two 16x16 tiles over 32 key positions ----------
    v8f s0 = {}, s1 = {};
    {
      const _Float16* kr0 = Kk + base + (size_t)(k0 + ln16) * HD_ + 16 * hi;
      const _Float16* kr1 = Kk + base + (size_t)(k0 + 16 + ln16) * HD_ + 16 * hi;
#pragma unroll
      for (int s = 0; s < 2; ++s) s0 = wmma_f16(aQ[s], ld_bfrag(kr0 + s * 32), s0);
#pragma unroll
      for (int s = 0; s < 2; ++s) s1 = wmma_f16(aQ[s], ld_bfrag(kr1 + s * 32), s1);
    }

    // ---- causal mask + online softmax ---------------------------------
#pragma unroll
    for (int r = 0; r < 8; ++r) {
      int qrow = q0 + r + hi * 8;
      int kc0  = k0 + ln16;
      float e0 = (kc0      <= qrow) ? s0[r] : -1e30f;
      float e1 = (kc0 + 16 <= qrow) ? s1[r] : -1e30f;
      float cm = fmaxf(e0, e1);
      cm = fmaxf(cm, __shfl_xor(cm, 1, 32));
      cm = fmaxf(cm, __shfl_xor(cm, 2, 32));
      cm = fmaxf(cm, __shfl_xor(cm, 4, 32));
      cm = fmaxf(cm, __shfl_xor(cm, 8, 32));
      float mnew  = fmaxf(mst[r], cm);
      float alpha = __expf(mst[r] - mnew);
      float p0 = __expf(e0 - mnew);
      float p1 = __expf(e1 - mnew);
      float rs = p0 + p1;
      rs += __shfl_xor(rs, 1, 32);
      rs += __shfl_xor(rs, 2, 32);
      rs += __shfl_xor(rs, 4, 32);
      rs += __shfl_xor(rs, 8, 32);
      lst[r] = lst[r] * alpha + rs;
      mst[r] = mnew;
#pragma unroll
      for (int nt = 0; nt < 4; ++nt) o[nt][r] *= alpha;
      Ps[w][r + hi * 8][ln16]      = (_Float16)p0;
      Ps[w][r + hi * 8][16 + ln16] = (_Float16)p1;
    }
    asm volatile("s_wait_dscnt 0" ::: "memory");   // intra-wave LDS RAW

    v16h aP = ld_afrag(&Ps[w][ln16][16 * hi], &Ps[w][ln16][16 * hi + 8]);

    // ---- O += P @ V  (V^T rows are k-contiguous) -----------------------
#pragma unroll
    for (int nt = 0; nt < 4; ++nt) {
      v16h bv = ld_bfrag(Vt + base + (size_t)(nt * 16 + ln16) * T_ + k0 + 16 * hi);
      o[nt] = wmma_f16(aP, bv, o[nt]);
    }
  }

  // ---- normalize + store f16 [B,T,C] (A-ready for out projection) ------
#pragma unroll
  for (int r = 0; r < 8; ++r) {
    float inv = 1.0f / lst[r];
    int qrow = q0 + r + hi * 8;
    size_t obase = ((size_t)(b * T_ + qrow)) * C_ + h * HD_;
#pragma unroll
    for (int nt = 0; nt < 4; ++nt)
      Oh[obase + nt * 16 + ln16] = (_Float16)(o[nt][r] * inv);
  }
}

// ---------------------------------------------------------------------------
// Kernel 3: output projection.  Oh[8192x768] (f16) @ Wpt[768x768]^T + b -> f32
// ---------------------------------------------------------------------------
__global__ __launch_bounds__(128)
void proj_gemm_kernel(const _Float16* __restrict__ A,
                      const _Float16* __restrict__ Wt,
                      const float* __restrict__ bias, float* __restrict__ Out) {
  const int lane = threadIdx.x & 31;
  const int wm   = threadIdx.x >> 5;
  const int ln16 = lane & 15;
  const int hi   = lane >> 4;
  const int n0   = blockIdx.x * 64;
  const int m0   = blockIdx.y * 64;

  const _Float16* arow = A + (size_t)(m0 + wm * 16 + ln16) * C_;
  v8f acc[4] = {};

  for (int k0 = 0; k0 < C_; k0 += 32) {
    v16h a = ld_afrag(arow + k0 + 16 * hi, arow + k0 + 16 * hi + 8);
#pragma unroll
    for (int nt = 0; nt < 4; ++nt) {
      v16h b = ld_bfrag(Wt + (size_t)(n0 + nt * 16 + ln16) * C_ + k0 + 16 * hi);
      acc[nt] = wmma_f16(a, b, acc[nt]);
    }
  }

#pragma unroll
  for (int nt = 0; nt < 4; ++nt) {
#pragma unroll
    for (int r = 0; r < 8; ++r) {
      int m = m0 + wm * 16 + r + hi * 8;
      int c = n0 + nt * 16 + ln16;
      Out[(size_t)m * C_ + c] = acc[nt][r] + bias[c];
    }
  }
}

// ---------------------------------------------------------------------------
// Launch.  Inputs: {x, mask, W_qkv, b_qkv, W_proj, b_proj}; mask is causal
// tril, applied analytically in attn_kernel.
// ---------------------------------------------------------------------------
extern "C" void kernel_launch(void* const* d_in, const int* in_sizes, int n_in,
                              void* d_out, int out_size, void* d_ws, size_t ws_size,
                              hipStream_t stream) {
  (void)in_sizes; (void)n_in; (void)out_size; (void)ws_size;

  const float* x     = (const float*)d_in[0];
  const float* Wqkv  = (const float*)d_in[2];
  const float* bqkv  = (const float*)d_in[3];
  const float* Wproj = (const float*)d_in[4];
  const float* bproj = (const float*)d_in[5];
  float* out = (float*)d_out;

  const size_t per = (size_t)B_ * H_ * T_ * HD_;    // 6,291,456 = B*T*C
  _Float16* q    = (_Float16*)d_ws;
  _Float16* k    = q + per;
  _Float16* vt   = k + per;                          // [B,H,HD,T]
  _Float16* oh   = vt + per;                         // [B,T,C]
  _Float16* x16  = oh + per;                         // [B*T, C]
  _Float16* wqt  = x16 + per;                        // [3C, C]
  _Float16* wpt  = wqt + (size_t)N_QKV * C_;         // [C, C]

  // prep: f16 convert + tiled weight transpose
  cvt_f32_f16_kernel<<<(int)((per / 4 + 255) / 256), 256, 0, stream>>>(
      x, x16, (int)(per / 4));
  transpose_cvt_kernel<<<dim3(N_QKV / 32, C_ / 32), 256, 0, stream>>>(
      Wqkv, wqt, C_, N_QKV);
  transpose_cvt_kernel<<<dim3(C_ / 32, C_ / 32), 256, 0, stream>>>(
      Wproj, wpt, C_, C_);

  dim3 blk(128);
  qkv_gemm_kernel<<<dim3(N_QKV / 64, (B_ * T_) / 64), blk, 0, stream>>>(
      x16, wqt, bqkv, q, k, vt);
  attn_kernel<<<dim3((B_ * H_ * (T_ / 16)) / 4), blk, 0, stream>>>(
      q, k, vt, oh);
  proj_gemm_kernel<<<dim3(C_ / 64, (B_ * T_) / 64), blk, 0, stream>>>(
      oh, wpt, bproj, out);
}
